// NNPhotonicMamba_35828617183350
// MI455X (gfx1250) — compile-verified, matrix operand order checked by reading, not run on hardware
//
#include <hip/hip_runtime.h>
#include <hip/hip_bf16.h>

// ---------------------------------------------------------------------------
// Photonic-Mamba forward for MI455X (gfx1250, wave32, WMMA).
//  * iso-linear layers collapsed to single bf16 GEMMs (exact for A_MAX=1,
//    ABS_AMIN=0):  x@Wp.T + (1-x)@Wn.T + bp - c == x@(Wp-Wn).T + rowsum(Wn)+bp-c
//  * GEMMs: v_wmma_f32_16x16x32_bf16, f32 accumulate. K templated so B-tile
//    strides become load immediates; A double-buffered, B loads batched into
//    one clause per k-step; register budget kept < ~100 VGPRs (no spills).
//  * selective scan: chunked associative scan over L (8 chunks of 64) for 8x
//    occupancy and 4x shorter critical path vs. fully sequential.
// ---------------------------------------------------------------------------

typedef __attribute__((ext_vector_type(16))) __bf16 v16bf;
typedef __attribute__((ext_vector_type(8)))  float  v8f;

#define DM      1024       // D_MODEL
#define DN      2048       // D_INNER
#define NS      16         // N_STATE
#define DTR     64         // DT_RANK
#define LSEQ    512
#define BSZ     2
#define BL      (BSZ*LSEQ) // 1024 rows
#define NC      8          // scan chunks
#define CHUNK   (LSEQ/NC)  // 64

union Frag { v16bf v; uint4 q[2]; };

// ---------------- weight prep: Wc = Wp - Wn (bf16), rowsum = sum(Wn) --------
__global__ void combine_weights_k(const float* __restrict__ Wp,
                                  const float* __restrict__ Wn,
                                  __hip_bfloat16* __restrict__ Wc,
                                  float* __restrict__ rowsum, int K) {
  const int row = blockIdx.x;
  const float* wp = Wp + (size_t)row * K;
  const float* wn = Wn + (size_t)row * K;
  __hip_bfloat16* wc = Wc + (size_t)row * K;
  float s = 0.f;
  for (int k = threadIdx.x; k < K; k += 256) {
    float p = wp[k], n = wn[k];
    wc[k] = __float2bfloat16(p - n);
    s += n;
  }
  __shared__ float red[256];
  red[threadIdx.x] = s;
  __syncthreads();
  for (int off = 128; off > 0; off >>= 1) {
    if (threadIdx.x < off) red[threadIdx.x] += red[threadIdx.x + off];
    __syncthreads();
  }
  if (threadIdx.x == 0) rowsum[row] = red[0];
}

// bias = rowsum + add1 - sub1 - sub2 + add2 (null pointers skipped)
__global__ void make_bias_k(const float* __restrict__ rowsum,
                            const float* __restrict__ add1,
                            const float* __restrict__ sub1,
                            const float* __restrict__ sub2,
                            const float* __restrict__ add2,
                            float* __restrict__ out, int n) {
  int i = blockIdx.x * 256 + threadIdx.x;
  if (i >= n) return;
  float v = rowsum[i];
  if (add1) v += add1[i];
  if (sub1) v -= sub1[i];
  if (sub2) v -= sub2[i];
  if (add2) v += add2[i];
  out[i] = v;
}

// depthwise conv weights: wc = wp-wn ; bias = sum(wn) + bc' - cc
__global__ void conv_combine_k(const float* __restrict__ wp,
                               const float* __restrict__ wn,
                               const float* __restrict__ bcp,
                               const float* __restrict__ cc,
                               float* __restrict__ wc, float* __restrict__ bc) {
  int d = blockIdx.x * 256 + threadIdx.x;
  if (d >= DN) return;
  float s = 0.f;
#pragma unroll
  for (int t = 0; t < 4; ++t) {
    float p = wp[d * 4 + t], n = wn[d * 4 + t];
    wc[d * 4 + t] = p - n;
    s += n;
  }
  bc[d] = s + bcp[d] - cc[d];
}

// strided f32 -> bf16 convert
__global__ void cvt_bf16_k(const float* __restrict__ src,
                           __hip_bfloat16* __restrict__ dst,
                           int rows, int cols, int sld, int dld) {
  int i = blockIdx.x * 256 + threadIdx.x;
  if (i >= rows * cols) return;
  int r = i / cols, c = i % cols;
  dst[(size_t)r * dld + c] = __float2bfloat16(src[(size_t)r * sld + c]);
}

// ---------------- WMMA GEMM: C[BLxN] = A[BLxK] (bf16) @ W[NxK]^T + bias -----
// block = 256 threads = 8 waves; wave w handles M-tile (blockIdx.y*8+w),
// NT 16-col N-tiles starting at blockIdx.x*NT. K % 64 == 0, compile-time.
// MODE 0/3: plain (ldc)   MODE 1: in-proj split -> clip(x) / silu(z)
// MODE 2: softplus
template <int K, int NT, int MODE>
__global__ __launch_bounds__(256)
void gemm_bf16_k(const __hip_bfloat16* __restrict__ A,
                 const __hip_bfloat16* __restrict__ W,
                 const float* __restrict__ bias,
                 float* __restrict__ out0,
                 float* __restrict__ out1,
                 int ldc) {
  const int lane  = threadIdx.x & 31;
  const int wave  = threadIdx.x >> 5;
  const int laneL = lane & 15;                 // M index (A) / N index (B)
  const int koff  = (lane & 16) ? 8 : 0;       // ISA 16-bit fragment layout
  const int rowbase = (blockIdx.y * 8 + wave) * 16;
  const int ntile0  = blockIdx.x * NT;

  const uint4* arow  = reinterpret_cast<const uint4*>(A + (size_t)(rowbase + laneL) * K);
  const uint4* wbase = reinterpret_cast<const uint4*>(W + (size_t)(ntile0 * 16 + laneL) * K);

  v8f acc[NT];
  const v8f zero = {0.f, 0.f, 0.f, 0.f, 0.f, 0.f, 0.f, 0.f};
#pragma unroll
  for (int t = 0; t < NT; ++t) acc[t] = zero;

  auto loadA = [&](Frag& a, int k0) {
    a.q[0] = arow[(k0 + koff) >> 3];
    a.q[1] = arow[(k0 + koff + 16) >> 3];
  };
  // one clause of 2*NT b128 loads; tile stride t*16*K folds into immediates
  auto loadB = [&](Frag (&b)[NT], int k0) {
    const int base = (k0 + koff) >> 3;
#pragma unroll
    for (int t = 0; t < NT; ++t) {
      b[t].q[0] = wbase[base + t * (2 * K)];
      b[t].q[1] = wbase[base + t * (2 * K) + 2];
    }
  };
  auto mma = [&](Frag& a, Frag (&b)[NT]) {
#pragma unroll
    for (int t = 0; t < NT; ++t)
      acc[t] = __builtin_amdgcn_wmma_f32_16x16x32_bf16(
          false, a.v, false, b[t].v, (short)0, acc[t], false, false);
  };

  // pipeline: double-buffer A only (B buffer reused -> no scratch spills)
  Frag a0, a1, b[NT];
  loadA(a0, 0);
  for (int k0 = 0; k0 < K; k0 += 64) {
    loadB(b, k0);
    loadA(a1, k0 + 32);
    __builtin_prefetch(arow + ((k0 + 128 + koff) >> 3), 0, 2);  // speculative
    mma(a0, b);
    loadB(b, k0 + 32);
    if (k0 + 64 < K) loadA(a0, k0 + 64);
    mma(a1, b);
  }

  const int rbase = rowbase + ((lane & 16) ? 8 : 0);
#pragma unroll
  for (int t = 0; t < NT; ++t) {
    const int n = (ntile0 + t) * 16 + laneL;
    const float bv = bias[n];
#pragma unroll
    for (int r = 0; r < 8; ++r) {
      const int row = rbase + r;
      float v = acc[t][r] + bv;
      if (MODE == 1) {
        if (n < DN) {
          out0[(size_t)row * DN + n] = fminf(fmaxf(v, 0.f), 1.f);      // x, clipped
        } else {
          out1[(size_t)row * DN + (n - DN)] = v / (1.f + __expf(-v));  // silu(z)
        }
      } else if (MODE == 2) {
        out0[(size_t)row * ldc + n] = (v > 20.f) ? v : log1pf(__expf(v)); // softplus
      } else {
        out0[(size_t)row * ldc + n] = v;
      }
    }
  }
}

// ---------------- causal depthwise conv (k=4) + silu + clip -----------------
__global__ void conv_silu_k(const float* __restrict__ x,
                            const float* __restrict__ wc,
                            const float* __restrict__ bc,
                            float* __restrict__ xo32,
                            __hip_bfloat16* __restrict__ xo16) {
  int idx = blockIdx.x * 256 + threadIdx.x;   // over BL * DN
  if (idx >= BL * DN) return;
  const int d  = idx % DN;
  const int rl = idx / DN;
  const int b  = rl / LSEQ;
  const int l  = rl % LSEQ;
  float acc = bc[d];
  const float* w = wc + d * 4;
#pragma unroll
  for (int t = 0; t < 4; ++t) {
    const int lt = l - 3 + t;
    const float xv = (lt >= 0) ? x[(size_t)(b * LSEQ + lt) * DN + d] : 0.f;
    acc += w[t] * xv;
  }
  float s = acc / (1.f + __expf(-acc));
  s = fminf(fmaxf(s, 0.f), 1.f);
  xo32[idx] = s;
  xo16[idx] = __float2bfloat16(s);
}

// ---------------- chunked selective scan -----------------------------------
// Recurrence h <- a*h + u is associative per state n. Split L into NC chunks:
// phase 1 computes each chunk's (prod a, local Bp, local Bd) from h=0;
// phase 2 folds predecessor chunk summaries (uniform per block) and replays
// the chunk to emit y. Summary layout: [b][c][n][d] (d-coalesced).
__global__ void scan_phase1_k(const float* __restrict__ delta,
                              const float* __restrict__ xc,
                              const float* __restrict__ ynn,   // BL x 96
                              const float* __restrict__ A_log,
                              const float* __restrict__ cx,
                              float* __restrict__ P,
                              float* __restrict__ Bp,
                              float* __restrict__ Bd) {
  const int db = blockIdx.x & 7;
  const int c  = (blockIdx.x >> 3) & 7;
  const int b  = blockIdx.x >> 6;
  const int d  = db * 256 + threadIdx.x;
  float Ad[NS], cB[NS], p[NS], bp[NS], bd[NS];
#pragma unroll
  for (int n = 0; n < NS; ++n) {
    Ad[n] = __expf(A_log[d * NS + n]);
    cB[n] = cx[DTR + n];
    p[n] = 1.f; bp[n] = 0.f; bd[n] = 0.f;
  }
  for (int l = c * CHUNK; l < c * CHUNK + CHUNK; ++l) {
    const size_t row = (size_t)b * LSEQ + l;
    const float dl = delta[row * DN + d];
    const float xv = xc[row * DN + d];
    const float* yr = ynn + row * 96;
#pragma unroll
    for (int n = 0; n < NS; ++n) {
      const float a = __expf(-dl * Ad[n]);
      p[n]  *= a;
      bp[n]  = a * bp[n] + dl * yr[DTR + n] * xv;
      bd[n]  = a * bd[n] + dl * xv * cB[n];
    }
  }
  const size_t base = (size_t)(b * NC + c) * NS * DN + d;
#pragma unroll
  for (int n = 0; n < NS; ++n) {
    P [base + (size_t)n * DN] = p[n];
    Bp[base + (size_t)n * DN] = bp[n];
    Bd[base + (size_t)n * DN] = bd[n];
  }
}

__global__ void scan_phase2_k(const float* __restrict__ delta,
                              const float* __restrict__ xc,
                              const float* __restrict__ ynn,
                              const float* __restrict__ zs,    // silu(z)
                              const float* __restrict__ A_log,
                              const float* __restrict__ cx,
                              const float* __restrict__ Dpos,
                              const float* __restrict__ Dneg,
                              const float* __restrict__ cD,
                              const float* __restrict__ P,
                              const float* __restrict__ Bp,
                              const float* __restrict__ Bd,
                              __hip_bfloat16* __restrict__ ybf) {
  const int db = blockIdx.x & 7;
  const int c  = (blockIdx.x >> 3) & 7;   // uniform per block
  const int b  = blockIdx.x >> 6;
  const int d  = db * 256 + threadIdx.x;
  float Ad[NS], cB[NS], cC[NS], hp[NS], hd[NS];
#pragma unroll
  for (int n = 0; n < NS; ++n) {
    Ad[n] = __expf(A_log[d * NS + n]);
    cB[n] = cx[DTR + n];
    cC[n] = cx[DTR + NS + n];
    hp[n] = 0.f; hd[n] = 0.f;
  }
  // fold predecessor chunk summaries (c <= 7 iterations, uniform)
  for (int i = 0; i < c; ++i) {
    const size_t base = (size_t)(b * NC + i) * NS * DN + d;
#pragma unroll
    for (int n = 0; n < NS; ++n) {
      const float pi = P[base + (size_t)n * DN];
      hp[n] = pi * hp[n] + Bp[base + (size_t)n * DN];
      hd[n] = pi * hd[n] + Bd[base + (size_t)n * DN];
    }
  }
  const float dp = Dpos[d], dng = Dneg[d], cd = cD[d];
  for (int l = c * CHUNK; l < c * CHUNK + CHUNK; ++l) {
    const size_t row = (size_t)b * LSEQ + l;
    const float dl = delta[row * DN + d];
    const float xv = xc[row * DN + d];
    const float* yr = ynn + row * 96;
    float yp = 0.f, y1 = 0.f, y2 = 0.f, y3 = 0.f;
#pragma unroll
    for (int n = 0; n < NS; ++n) {
      const float Bn = yr[DTR + n];
      const float Cn = yr[DTR + NS + n];
      const float a  = __expf(-dl * Ad[n]);
      hp[n] = a * hp[n] + dl * Bn * xv;
      hd[n] = a * hd[n] + dl * xv * cB[n];
      yp += Cn * hp[n];
      y1 += Cn * hd[n];
      y2 += cC[n] * hp[n];
      y3 += cC[n] * hd[n];
    }
    float y = yp - (y1 + y2 - y3);
    y += dp * xv + dng * (1.f - xv) - cd;          // D skip (iso form)
    y = fminf(fmaxf(y, 0.f), 1.f);
    y *= zs[row * DN + d];
    ybf[row * DN + d] = __float2bfloat16(y);
  }
}

// ---------------------------------------------------------------------------
extern "C" void kernel_launch(void* const* d_in, const int* in_sizes, int n_in,
                              void* d_out, int out_size, void* d_ws, size_t ws_size,
                              hipStream_t stream) {
  (void)in_sizes; (void)n_in; (void)out_size; (void)ws_size;
  const float* hs      = (const float*)d_in[0];
  const float* Win_p   = (const float*)d_in[1];
  const float* Win_n   = (const float*)d_in[2];
  const float* bin_p   = (const float*)d_in[3];
  const float* cin     = (const float*)d_in[4];
  const float* wc_p    = (const float*)d_in[5];
  const float* wc_n    = (const float*)d_in[6];
  const float* bc_p    = (const float*)d_in[7];
  const float* cc      = (const float*)d_in[8];
  const float* Wx_p    = (const float*)d_in[9];
  const float* Wx_n    = (const float*)d_in[10];
  const float* bx_p    = (const float*)d_in[11];
  const float* cx      = (const float*)d_in[12];
  const float* Wdt_p   = (const float*)d_in[13];
  const float* Wdt_n   = (const float*)d_in[14];
  const float* bdt_p   = (const float*)d_in[15];
  const float* cdt     = (const float*)d_in[16];
  const float* dt_bias = (const float*)d_in[17];
  const float* dt_cc   = (const float*)d_in[18];
  const float* Wo_p    = (const float*)d_in[19];
  const float* Wo_n    = (const float*)d_in[20];
  const float* bo_p    = (const float*)d_in[21];
  const float* co      = (const float*)d_in[22];
  const float* A_log   = (const float*)d_in[23];
  const float* Dpos    = (const float*)d_in[24];
  const float* Dneg    = (const float*)d_in[25];
  const float* cD      = (const float*)d_in[26];

  char* ws = (char*)d_ws;
  size_t off = 0;
  auto alloc = [&](size_t bytes) -> char* {
    char* p = ws + off;
    off += (bytes + 255) & ~(size_t)255;
    return p;
  };

  __hip_bfloat16* Wc_in  = (__hip_bfloat16*)alloc((size_t)(2 * DN) * DM * 2);
  __hip_bfloat16* Wc_x   = (__hip_bfloat16*)alloc((size_t)96 * DN * 2);
  __hip_bfloat16* Wc_dt  = (__hip_bfloat16*)alloc((size_t)DN * DTR * 2);
  __hip_bfloat16* Wc_out = (__hip_bfloat16*)alloc((size_t)DM * DN * 2);
  float* rs_in  = (float*)alloc((size_t)(2 * DN) * 4);
  float* rs_x   = (float*)alloc(96 * 4);
  float* rs_dt  = (float*)alloc((size_t)DN * 4);
  float* rs_out = (float*)alloc((size_t)DM * 4);
  float* b_in   = (float*)alloc((size_t)(2 * DN) * 4);
  float* b_x    = (float*)alloc(96 * 4);
  float* b_dt   = (float*)alloc((size_t)DN * 4);
  float* b_out  = (float*)alloc((size_t)DM * 4);
  float* wconv  = (float*)alloc((size_t)DN * 4 * 4);
  float* bconv  = (float*)alloc((size_t)DN * 4);
  __hip_bfloat16* Xh   = (__hip_bfloat16*)alloc((size_t)BL * DM * 2);
  float* x_raw  = (float*)alloc((size_t)BL * DN * 4);
  float* zs     = (float*)alloc((size_t)BL * DN * 4);
  float* xc32   = (float*)alloc((size_t)BL * DN * 4);
  __hip_bfloat16* xc16 = (__hip_bfloat16*)alloc((size_t)BL * DN * 2);
  float* ynn    = (float*)alloc((size_t)BL * 96 * 4);
  __hip_bfloat16* dtb  = (__hip_bfloat16*)alloc((size_t)BL * DTR * 2);
  float* delta  = (float*)alloc((size_t)BL * DN * 4);
  __hip_bfloat16* ybf  = (__hip_bfloat16*)alloc((size_t)BL * DN * 2);
  float* Pbuf   = (float*)alloc((size_t)BSZ * NC * NS * DN * 4);
  float* Bpbuf  = (float*)alloc((size_t)BSZ * NC * NS * DN * 4);
  float* Bdbuf  = (float*)alloc((size_t)BSZ * NC * NS * DN * 4);

  // --- weight / bias prep ---
  combine_weights_k<<<2 * DN, 256, 0, stream>>>(Win_p, Win_n, Wc_in, rs_in, DM);
  combine_weights_k<<<96,     256, 0, stream>>>(Wx_p,  Wx_n,  Wc_x,  rs_x,  DN);
  combine_weights_k<<<DN,     256, 0, stream>>>(Wdt_p, Wdt_n, Wc_dt, rs_dt, DTR);
  combine_weights_k<<<DM,     256, 0, stream>>>(Wo_p,  Wo_n,  Wc_out, rs_out, DN);
  make_bias_k<<<16, 256, 0, stream>>>(rs_in,  bin_p, cin,  nullptr, nullptr, b_in, 2 * DN);
  make_bias_k<<<1,  256, 0, stream>>>(rs_x,   bx_p,  nullptr, nullptr, nullptr, b_x, 96);
  make_bias_k<<<8,  256, 0, stream>>>(rs_dt,  bdt_p, cdt,  dt_cc, dt_bias, b_dt, DN);
  make_bias_k<<<4,  256, 0, stream>>>(rs_out, bo_p,  co,   nullptr, nullptr, b_out, DM);
  conv_combine_k<<<8, 256, 0, stream>>>(wc_p, wc_n, bc_p, cc, wconv, bconv);
  cvt_bf16_k<<<(BL * DM) / 256, 256, 0, stream>>>(hs, Xh, BL, DM, DM, DM);

  const dim3 blk(256, 1, 1);
  // --- in-proj GEMM (N=4096, K=1024), fused clip(x) / silu(z) split ---
  gemm_bf16_k<DM, 4, 1><<<dim3((2 * DN) / 64, 8), blk, 0, stream>>>(
      Xh, Wc_in, b_in, x_raw, zs, 2 * DN);
  // --- causal depthwise conv + silu + clip ---
  conv_silu_k<<<(BL * DN) / 256, 256, 0, stream>>>(x_raw, wconv, bconv, xc32, xc16);
  // --- x-proj GEMM (N=96, K=2048) ---
  gemm_bf16_k<DN, 2, 0><<<dim3(96 / 32, 8), blk, 0, stream>>>(
      xc16, Wc_x, b_x, ynn, nullptr, 96);
  // --- dt slice -> bf16, dt GEMM (N=2048, K=64) with fused softplus ---
  cvt_bf16_k<<<(BL * DTR) / 256, 256, 0, stream>>>(ynn, dtb, BL, DTR, 96, DTR);
  gemm_bf16_k<DTR, 4, 2><<<dim3(DN / 64, 8), blk, 0, stream>>>(
      dtb, Wc_dt, b_dt, delta, nullptr, DN);
  // --- chunked selective scan (2 phases), fused epilogue ---
  scan_phase1_k<<<BSZ * NC * (DN / 256), 256, 0, stream>>>(
      delta, xc32, ynn, A_log, cx, Pbuf, Bpbuf, Bdbuf);
  scan_phase2_k<<<BSZ * NC * (DN / 256), 256, 0, stream>>>(
      delta, xc32, ynn, zs, A_log, cx, Dpos, Dneg, cD, Pbuf, Bpbuf, Bdbuf, ybf);
  // --- out-proj GEMM (N=1024, K=2048) -> d_out (f32) ---
  gemm_bf16_k<DN, 4, 3><<<dim3(DM / 64, 8), blk, 0, stream>>>(
      ybf, Wc_out, b_out, (float*)d_out, nullptr, DM);
}
